// PaLM_28054726377506
// MI455X (gfx1250) — matrix operand order repaired and checked by reading, not compiled, not run on hardware
//
#include <hip/hip_runtime.h>

// ============================================================================
// PaLM forward on MI455X (gfx1250, wave32, WMMA).
// Precision: fp32 residual stream; all GEMMs run bf16 x bf16 -> f32 accumulate
// via v_wmma_f32_16x16x32_bf16 (fp32 WMMA is only 16x16x4 -> 8x less K/instr).
// Weights are converted f32->bf16 while staging tiles into LDS. The bf16 A
// tiles are staged with GLOBAL_LOAD_ASYNC_TO_LDS_B128 (ASYNCcnt path, inline
// asm per CDNA5 ISA). ws usage ~288 MB (documented below).
// ============================================================================

typedef unsigned short u16;
typedef unsigned int   u32;

typedef __bf16 v16bf __attribute__((ext_vector_type(16)));
typedef float  v8f   __attribute__((ext_vector_type(8)));

union Frag { v16bf f; uint4 q[2]; };
static_assert(sizeof(Frag) == 32, "frag size");

// ---- model constants -------------------------------------------------------
#define NDIM     2048
#define NHEADS   16
#define DHEAD    128
#define FFIN     8192
#define NFUSED   18688          // 16*128 q + 128 k + 128 v + 2*8192 ff
#define NVOCAB   32000
#define NB       2
#define NSEQ     2048
#define NROWS    4096           // B*N tokens
#define OFF_K    2048           // col offset of k in proj
#define OFF_V    2176
#define OFF_FF   2304

__device__ __forceinline__ u16 f32_to_bf16(float f) {
  u32 u = __float_as_uint(f);
  u = (u + 0x7FFFu + ((u >> 16) & 1u)) >> 16;        // round-to-nearest-even
  return (u16)u;
}
__device__ __forceinline__ float bf16_to_f32(u16 h) {
  return __uint_as_float(((u32)h) << 16);
}
__device__ __forceinline__ float rowmax16(float x) {
  #pragma unroll
  for (int m = 1; m < 16; m <<= 1) x = fmaxf(x, __shfl_xor(x, m, 32));
  return x;
}
__device__ __forceinline__ float rowsum16(float x) {
  #pragma unroll
  for (int m = 1; m < 16; m <<= 1) x += __shfl_xor(x, m, 32);
  return x;
}

// ---- CDNA5 async memory->LDS (per-lane 16B), tracked by ASYNCcnt -----------
// VDST = LDS byte offset (generic LDS pointer low 32 bits = LDS_ADDR per
// aperture rules), VADDR = 64-bit global address, no SADDR (GV mode).
__device__ __forceinline__ void async_load_b128(u32 lds_off, const void* gaddr) {
  asm volatile("global_load_async_to_lds_b128 %0, %1, off"
               :: "v"(lds_off), "v"(gaddr) : "memory");
}
__device__ __forceinline__ void wait_async0() {
  asm volatile("s_wait_asynccnt 0x0" ::: "memory");
}

// ============================================================================
// Embedding gather: x[row,:] = emb[tokens[row],:]
// ============================================================================
__global__ __launch_bounds__(256) void k_embed(const int* __restrict__ tokens,
                                               const float* __restrict__ emb,
                                               float* __restrict__ x) {
  const int row = blockIdx.x;
  const int tok = tokens[row];
  const float4* src = (const float4*)(emb + (size_t)tok * NDIM);
  float4*       dst = (float4*)(x + (size_t)row * NDIM);
  dst[threadIdx.x]       = src[threadIdx.x];
  dst[threadIdx.x + 256] = src[threadIdx.x + 256];
}

// ============================================================================
// LayerNorm (beta=0) -> bf16 output. One 256-thread block per row of 2048.
// ============================================================================
__global__ __launch_bounds__(256) void k_ln(const float* __restrict__ x,
                                            const float* __restrict__ gamma,
                                            u16* __restrict__ out) {
  __shared__ float red[256];
  const int row = blockIdx.x, tid = threadIdx.x;
  const float* xr = x + (size_t)row * NDIM;
  float vbuf[8], s = 0.f, s2 = 0.f;
  #pragma unroll
  for (int e = 0; e < 8; ++e) {
    float v = xr[tid * 8 + e]; vbuf[e] = v; s += v; s2 += v * v;
  }
  red[tid] = s; __syncthreads();
  for (int o = 128; o > 0; o >>= 1) { if (tid < o) red[tid] += red[tid + o]; __syncthreads(); }
  const float mean = red[0] * (1.f / NDIM);
  __syncthreads();
  red[tid] = s2; __syncthreads();
  for (int o = 128; o > 0; o >>= 1) { if (tid < o) red[tid] += red[tid + o]; __syncthreads(); }
  const float var  = red[0] * (1.f / NDIM) - mean * mean;
  const float rstd = rsqrtf(var + 1e-5f);
  #pragma unroll
  for (int e = 0; e < 8; ++e)
    out[(size_t)row * NDIM + tid * 8 + e] =
        f32_to_bf16((vbuf[e] - mean) * rstd * gamma[tid * 8 + e]);
}

// ============================================================================
// bf16 WMMA GEMM:  C[M,N] = A_bf16[M,K] * W_f32, W converted to bf16 in LDS.
//   B_TRANS=0 : W is [K,N] row-major (weights)
//   B_TRANS=1 : W is [N,K] row-major (tied-embedding logits: emb[vocab,dim])
//   OUT_BF16  : write bf16; ACCUM : C_f32 += (residual adds)
// Block tile 64x256, BK=32 (one WMMA K-step). 8 waves as 2x4, 32x64 per wave
// (8 WMMAs per 12 fragment loads). A tile staged via async-to-LDS (bf16 in
// memory, no conversion needed); B tile converted f32->bf16 and transposed.
// Fragment layouts per CDNA5 ISA 7.12.2:
//   A(16x32): lane g=l>>4,r=l&15 -> row r, K {g*8..g*8+7, g*8+16..g*8+23}
//   B(32x16): lane -> col r, K {g*8..g*8+7, g*8+16..g*8+23} -> LDS B stored transposed [n][k]
//   C(16x16): lane -> col r, rows {g*8+v}, v = VGPR index
// ============================================================================
#define BM 64
#define BN 256
#define BK 32
#define SB_LD 40   // padded k-stride of transposed B tile (bytes%16==0)

template <bool B_TRANS, bool OUT_BF16, bool ACCUM>
__global__ __launch_bounds__(256)
void k_gemm(const u16* __restrict__ A, const float* __restrict__ W,
            void* __restrict__ Cp, int M, int N, int K) {
  __shared__ __align__(16) u16 sA[BM * BK];        //  4 KB
  __shared__ __align__(16) u16 sB[BN * SB_LD];     // 20 KB

  const int tid  = threadIdx.x;
  const int lane = tid & 31;
  const int w    = tid >> 5;
  const int wm   = w & 1, wn = w >> 1;             // 2 x 4 wave grid
  const int m0   = blockIdx.y * BM;
  const int n0   = blockIdx.x * BN;
  const int r    = lane & 15, g = lane >> 4;

  v8f acc[2][4];
  #pragma unroll
  for (int i = 0; i < 2; ++i)
    #pragma unroll
    for (int j = 0; j < 4; ++j)
      #pragma unroll
      for (int e = 0; e < 8; ++e) acc[i][j][e] = 0.f;

  // A stager: 256 threads x 16B cover the 64x32 bf16 tile exactly.
  const int ar = tid >> 2, ac = (tid & 3) * 8;
  const u32 sa_off = (u32)(size_t)(sA + ar * BK + ac);   // LDS byte offset

  for (int kt = 0; kt < K; kt += BK) {
    // ---- stage A tile: async global -> LDS (ASYNCcnt) ----
    async_load_b128(sa_off, A + (size_t)(m0 + ar) * K + kt + ac);
    if (kt + BK < K)
      __builtin_prefetch(A + (size_t)(m0 + ar) * K + kt + BK + ac, 0, 0);

    // ---- stage B tile, f32 -> bf16, transposed to [n][k] ----
    if (B_TRANS) {
      const int n = tid;                              // one row per thread
      const float* src = W + (size_t)(n0 + n) * K + kt;
      #pragma unroll
      for (int j = 0; j < BK; j += 4) {
        float4 f = *(const float4*)(src + j);
        sB[n * SB_LD + j + 0] = f32_to_bf16(f.x);
        sB[n * SB_LD + j + 1] = f32_to_bf16(f.y);
        sB[n * SB_LD + j + 2] = f32_to_bf16(f.z);
        sB[n * SB_LD + j + 3] = f32_to_bf16(f.w);
      }
      if (kt + BK < K) __builtin_prefetch(src + BK, 0, 0);
    } else {
      const int k = tid >> 3, nb = (tid & 7) * 32;    // 32 cols per thread
      const float* src = W + (size_t)(kt + k) * N + n0 + nb;
      #pragma unroll
      for (int j = 0; j < 32; j += 4) {
        float4 f = *(const float4*)(src + j);
        sB[(nb + j + 0) * SB_LD + k] = f32_to_bf16(f.x);
        sB[(nb + j + 1) * SB_LD + k] = f32_to_bf16(f.y);
        sB[(nb + j + 2) * SB_LD + k] = f32_to_bf16(f.z);
        sB[(nb + j + 3) * SB_LD + k] = f32_to_bf16(f.w);
      }
      if (kt + BK < K) __builtin_prefetch(src + (size_t)BK * N, 0, 0);
    }
    wait_async0();            // A tile landed in LDS
    __syncthreads();

    // ---- fragments + WMMA (2x4 tiles of 16x16 per wave) ----
    Frag af[2], bfr[4];
    #pragma unroll
    for (int mi = 0; mi < 2; ++mi) {
      const int rr = wm * 32 + mi * 16 + r;
      af[mi].q[0] = *(const uint4*)(sA + rr * BK + g * 8);
      af[mi].q[1] = *(const uint4*)(sA + rr * BK + g * 8 + 16);
    }
    #pragma unroll
    for (int ni = 0; ni < 4; ++ni) {
      const int cc = wn * 64 + ni * 16 + r;
      bfr[ni].q[0] = *(const uint4*)(sB + cc * SB_LD + g * 8);
      bfr[ni].q[1] = *(const uint4*)(sB + cc * SB_LD + g * 8 + 16);
    }
    #pragma unroll
    for (int mi = 0; mi < 2; ++mi)
      #pragma unroll
      for (int ni = 0; ni < 4; ++ni)
        acc[mi][ni] = __builtin_amdgcn_wmma_f32_16x16x32_bf16(
            false, af[mi].f, false, bfr[ni].f, (short)0, acc[mi][ni],
            false, false);
    __syncthreads();
  }

  // ---- epilogue ----
  #pragma unroll
  for (int mi = 0; mi < 2; ++mi)
    #pragma unroll
    for (int ni = 0; ni < 4; ++ni) {
      const int mb = m0 + wm * 32 + mi * 16 + g * 8;
      const int nn = n0 + wn * 64 + ni * 16 + r;
      #pragma unroll
      for (int v = 0; v < 8; ++v) {
        const size_t idx = (size_t)(mb + v) * N + nn;
        if (ACCUM)          ((float*)Cp)[idx] += acc[mi][ni][v];
        else if (OUT_BF16)  ((u16*)Cp)[idx]   = f32_to_bf16(acc[mi][ni][v]);
        else                ((float*)Cp)[idx] = acc[mi][ni][v];
      }
    }
}

// ============================================================================
// RoPE (in-place on bf16 proj): q heads (with 1/sqrt(dhead) scale) + shared k.
// grid (4096 rows, 17 units), 64 threads: lane d pairs (d, d+64).
// ============================================================================
__global__ __launch_bounds__(64) void k_rope(u16* __restrict__ proj) {
  const int d   = threadIdx.x;
  const int row = blockIdx.x;
  const int hh  = blockIdx.y;                 // 0..15 = q heads, 16 = k
  const int n   = row & (NSEQ - 1);           // position
  const float invf = __expf(-(float)d * 0.14391156f);   // ln(10000)/64
  float s, c;
  __sincosf((float)n * invf, &s, &c);
  u16* base = proj + (size_t)row * NFUSED + (hh < 16 ? hh * DHEAD : OFF_K);
  const float scl = (hh < 16) ? 0.08838834764831845f : 1.0f;  // dhead^-0.5
  const float x1 = bf16_to_f32(base[d]);
  const float x2 = bf16_to_f32(base[d + 64]);
  base[d]      = f32_to_bf16((x1 * c - x2 * s) * scl);
  base[d + 64] = f32_to_bf16((x2 * c + x1 * s) * scl);
}

// ============================================================================
// Flash attention (multi-query, causal), WMMA bf16.
// grid (16 q-blocks, 16 heads, 2 batches); 256 thr; wave owns 16 queries.
// S = Q*K^T : Q A-frags preloaded from global (roped, scaled); K^T B-frags
//             read straight from global (contiguous in head-dim).
// P*V       : V chunk staged transposed in LDS; P bounced through per-wave LDS
//             to convert C-layout -> A-fragment layout.
// ============================================================================
__global__ __launch_bounds__(256)
void k_attn(const u16* __restrict__ proj, u16* __restrict__ aout) {
  __shared__ __align__(16) u16 sV[DHEAD * 32];        // [d][key]   8 KB
  __shared__ __align__(16) u16 sP[8 * 16 * 32];       // per-wave P 8 KB

  const int tid = threadIdx.x, lane = tid & 31, w = tid >> 5;
  const int qb = blockIdx.x, h = blockIdx.y, b = blockIdx.z;
  const int r = lane & 15, g = lane >> 4;
  const int qn = qb * 128 + w * 16;                   // wave's first query
  const size_t bn = (size_t)b * NSEQ;

  // preload Q fragments (4 K-steps over head dim)
  Frag qf[4];
  {
    const size_t qrow = (bn + qn + r) * NFUSED + (size_t)h * DHEAD;
    #pragma unroll
    for (int kk = 0; kk < 4; ++kk) {
      qf[kk].q[0] = *(const uint4*)(proj + qrow + kk * 32 + g * 8);
      qf[kk].q[1] = *(const uint4*)(proj + qrow + kk * 32 + g * 8 + 16);
    }
  }

  v8f O[8];
  float m_[8], l_[8];
  #pragma unroll
  for (int dg = 0; dg < 8; ++dg)
    #pragma unroll
    for (int e = 0; e < 8; ++e) O[dg][e] = 0.f;
  #pragma unroll
  for (int v = 0; v < 8; ++v) { m_[v] = -1e30f; l_[v] = 0.f; }

  const int nchunks = (qb + 1) * 4;                   // 32 keys / chunk
  for (int ch = 0; ch < nchunks; ++ch) {
    const int j0 = ch * 32;

    // ---- stage V chunk transposed: sV[d][key] ----
    {
      const int key = tid >> 3, db = (tid & 7) * 16;
      const u16* src = proj + (bn + j0 + key) * NFUSED + OFF_V + db;
      const uint4 u0 = *(const uint4*)(src);
      const uint4 u1 = *(const uint4*)(src + 8);
      const u32 packed[8] = {u0.x, u0.y, u0.z, u0.w, u1.x, u1.y, u1.z, u1.w};
      #pragma unroll
      for (int e = 0; e < 8; ++e) {
        sV[(db + 2 * e) * 32 + key]     = (u16)(packed[e] & 0xFFFFu);
        sV[(db + 2 * e + 1) * 32 + key] = (u16)(packed[e] >> 16);
      }
    }
    __syncthreads();

    // ---- S = Q K^T for two 16-key tiles ----
    v8f s[2];
    #pragma unroll
    for (int jt = 0; jt < 2; ++jt)
      #pragma unroll
      for (int e = 0; e < 8; ++e) s[jt][e] = 0.f;
    #pragma unroll
    for (int jt = 0; jt < 2; ++jt) {
      const int key = j0 + jt * 16 + r;
      const size_t kb = (bn + key) * NFUSED + OFF_K;
      #pragma unroll
      for (int kk = 0; kk < 4; ++kk) {
        Frag kf;
        kf.q[0] = *(const uint4*)(proj + kb + kk * 32 + g * 8);
        kf.q[1] = *(const uint4*)(proj + kb + kk * 32 + g * 8 + 16);
        s[jt] = __builtin_amdgcn_wmma_f32_16x16x32_bf16(
            false, qf[kk].f, false, kf.f, (short)0, s[jt], false, false);
      }
    }

    // ---- online softmax (rows striped over 16 lanes; reduce via shfl) ----
    const int key0 = j0 + r, key1 = j0 + 16 + r;
    #pragma unroll
    for (int v = 0; v < 8; ++v) {
      const int qv = qn + g * 8 + v;
      if (key0 > qv) s[0][v] = -1e30f;
      if (key1 > qv) s[1][v] = -1e30f;
      const float rm = rowmax16(fmaxf(s[0][v], s[1][v]));
      const float mn = fmaxf(m_[v], rm);
      const float sc = __expf(m_[v] - mn);
      const float p0 = (key0 > qv) ? 0.f : __expf(s[0][v] - mn);
      const float p1 = (key1 > qv) ? 0.f : __expf(s[1][v] - mn);
      l_[v] = l_[v] * sc + rowsum16(p0 + p1);
      m_[v] = mn;
      #pragma unroll
      for (int dg = 0; dg < 8; ++dg) O[dg][v] *= sc;
      sP[(w * 16 + g * 8 + v) * 32 + r]      = f32_to_bf16(p0);
      sP[(w * 16 + g * 8 + v) * 32 + 16 + r] = f32_to_bf16(p1);
    }
    // per-wave LDS region; same-wave DS ops complete in order -> no barrier.

    // ---- O += P * V ----
    Frag pf;
    pf.q[0] = *(const uint4*)(sP + (w * 16 + r) * 32 + g * 8);
    pf.q[1] = *(const uint4*)(sP + (w * 16 + r) * 32 + g * 8 + 16);
    #pragma unroll
    for (int dg = 0; dg < 8; ++dg) {
      Frag vf;
      vf.q[0] = *(const uint4*)(sV + (dg * 16 + r) * 32 + g * 8);
      vf.q[1] = *(const uint4*)(sV + (dg * 16 + r) * 32 + g * 8 + 16);
      O[dg] = __builtin_amdgcn_wmma_f32_16x16x32_bf16(
          false, pf.f, false, vf.f, (short)0, O[dg], false, false);
    }
    __syncthreads();   // protect sV before next chunk staging
  }

  // ---- normalize + store (bf16, feeds attn-out GEMM) ----
  #pragma unroll
  for (int v = 0; v < 8; ++v) {
    const float inv = 1.f / l_[v];
    const size_t row = (bn + qn + g * 8 + v) * (size_t)NDIM + (size_t)h * DHEAD;
    #pragma unroll
    for (int dg = 0; dg < 8; ++dg)
      aout[row + dg * 16 + r] = f32_to_bf16(O[dg][v] * inv);
  }
}

// ============================================================================
// SwiGLU: ffact = silu(gate) * ff_x   (bf16 in proj -> bf16)
// ============================================================================
__global__ __launch_bounds__(256) void k_swiglu(const u16* __restrict__ proj,
                                                u16* __restrict__ ff) {
  const size_t i   = ((size_t)blockIdx.x * 256 + threadIdx.x) * 4;
  const size_t row = i >> 13;           // / 8192
  const size_t col = i & (FFIN - 1);
  const u16* p = proj + row * NFUSED + OFF_FF;
  #pragma unroll
  for (int e = 0; e < 4; ++e) {
    const float xg = bf16_to_f32(p[col + e]);
    const float gt = bf16_to_f32(p[FFIN + col + e]);
    const float sig = 1.f / (1.f + __expf(-gt));
    ff[row * FFIN + col + e] = f32_to_bf16(gt * sig * xg);
  }
}

// ============================================================================
// Launcher. Inputs: tokens, emb, gammas, gamma_final, w_fused, w_attn_out,
// w_ff_out. Output: logits f32 [B,N,VOCAB]. Workspace (~288 MB):
//   [0       ) x      f32  4096x2048   32 MB   residual stream
//   [32  MB  ) h      bf16 4096x2048   16 MB   LN output
//   [48  MB  ) proj   bf16 4096x18688 146 MB   fused projection (roped in place)
//   [194 MB  ) ain    bf16 4096x2048   16 MB   attention output
//   [210 MB  ) ffact  bf16 4096x8192   64 MB   swiglu activations
// ============================================================================
extern "C" void kernel_launch(void* const* d_in, const int* in_sizes, int n_in,
                              void* d_out, int out_size, void* d_ws,
                              size_t ws_size, hipStream_t stream) {
  (void)in_sizes; (void)n_in; (void)out_size; (void)ws_size;
  const int*   tokens  = (const int*)d_in[0];
  const float* emb     = (const float*)d_in[1];
  const float* gammas  = (const float*)d_in[2];
  const float* gfin    = (const float*)d_in[3];
  const float* w_fused = (const float*)d_in[4];
  const float* w_attn  = (const float*)d_in[5];
  const float* w_ff    = (const float*)d_in[6];

  char* ws = (char*)d_ws;
  float* x    = (float*)(ws);
  u16*   h    = (u16*)(ws + (size_t)33554432);
  u16*   proj = (u16*)(ws + (size_t)50331648);
  u16*   ain  = (u16*)(ws + (size_t)203423744);
  u16*   ffa  = (u16*)(ws + (size_t)220200960);

  k_embed<<<NROWS, 256, 0, stream>>>(tokens, emb, x);

  for (int l = 0; l < 2; ++l) {
    k_ln<<<NROWS, 256, 0, stream>>>(x, gammas + (size_t)l * NDIM, h);
    // proj = LN(x) @ w_fused  [4096 x 18688]
    k_gemm<false, true, false><<<dim3(NFUSED / BN, NROWS / BM), 256, 0, stream>>>(
        h, w_fused + (size_t)l * NDIM * NFUSED, proj, NROWS, NFUSED, NDIM);
    // rope on q (scaled) + shared k, in place
    k_rope<<<dim3(NROWS, NHEADS + 1), 64, 0, stream>>>(proj);
    // causal multi-query flash attention
    k_attn<<<dim3(NSEQ / 128, NHEADS, NB), 256, 0, stream>>>(proj, ain);
    // x += attn @ w_attn_out
    k_gemm<false, false, true><<<dim3(NDIM / BN, NROWS / BM), 256, 0, stream>>>(
        ain, w_attn + (size_t)l * NDIM * NDIM, x, NROWS, NDIM, NDIM);
    // swiglu, then x += ffact @ w_ff_out
    k_swiglu<<<(NROWS * FFIN) / 1024, 256, 0, stream>>>(proj, ffa);
    k_gemm<false, false, true><<<dim3(NDIM / BN, NROWS / BM), 256, 0, stream>>>(
        ffa, w_ff + (size_t)l * FFIN * NDIM, x, NROWS, NDIM, FFIN);
  }

  // final LN + tied logits: logits = LN(x) @ emb^T  (emb is [V,K] -> B_TRANS)
  k_ln<<<NROWS, 256, 0, stream>>>(x, gfin, h);
  k_gemm<true, false, false><<<dim3(NVOCAB / BN, NROWS / BM), 256, 0, stream>>>(
      h, emb, d_out, NROWS, NVOCAB, NDIM);
}
